// MambaBlock_85237920956534
// MI455X (gfx1250) — matrix operand order repaired and checked by reading, not compiled, not run on hardware
//
#include <hip/hip_runtime.h>
#include <hip/hip_bf16.h>

// ---------------------------------------------------------------------------
// MI455X (gfx1250) Mamba block.
//  - Big GEMMs: v_wmma_f32_16x16x32_bf16, 64x64 wave tiles (16 wmma / 8 frag
//    loads per K-step), DOUBLE-BUFFERED LDS fed by global_load_async_to_lds
//    (ASYNCcnt-tracked, zero VGPR staging -> no spills, real overlap).
//  - Diagonal SSM scan as 3-pass chunked parallel scan (16x shorter serial
//    dependency chains), rank-16 dt projection fused into the scan.
// ---------------------------------------------------------------------------

typedef __attribute__((ext_vector_type(16))) __bf16 v16bf;
typedef __attribute__((ext_vector_type(8)))  __bf16 v8bf;
typedef __attribute__((ext_vector_type(4)))  __bf16 v4bf;
typedef __attribute__((ext_vector_type(8)))  float  v8f;

#define DIMD   1024
#define RNK    16
#define BATCH  4
#define SEQ    4096
#define MROWS  (BATCH * SEQ)   // 16384
#define NCHUNK 16
#define CHLEN  (SEQ / NCHUNK)  // 256

// Async 16B global->LDS copy (CDNA5 GLOBAL_LOAD_ASYNC_TO_LDS_B128, ASYNCcnt).
// vdst = 32-bit LDS byte address, vaddr = 64-bit global address, saddr = off.
#define ASYNC_CP16(ldsp, gp)                                                  \
    asm volatile("global_load_async_to_lds_b128 %0, %1, off"                  \
                 :: "v"((unsigned)(uintptr_t)(ldsp)), "v"(gp) : "memory")
#define WAIT_ASYNC_8()  asm volatile("s_wait_asynccnt 0x8" ::: "memory")
#define WAIT_ASYNC_0()  asm volatile("s_wait_asynccnt 0x0" ::: "memory")

__device__ __forceinline__ float softplusf(float x) {
    return (x > 20.f) ? x : log1pf(__expf(x));
}
__device__ __forceinline__ float fast_sigmoid(float x) {
    return __builtin_amdgcn_rcpf(1.f + __expf(-x));  // v_rcp_f32, no IEEE divide
}

// ----------------------------- conversions ---------------------------------

__global__ void k_f32_to_bf16(const float* __restrict__ s, __bf16* __restrict__ d, int n) {
    int i = (blockIdx.x * blockDim.x + threadIdx.x) * 4;
    if (i >= n) return;
    float4 f = *(const float4*)(s + i);
    v4bf o;
    o[0] = (__bf16)f.x; o[1] = (__bf16)f.y; o[2] = (__bf16)f.z; o[3] = (__bf16)f.w;
    *(v4bf*)(d + i) = o;
}

// WT[n][k] = (bf16) W[k][n]   (W is [K,N] row-major)
__global__ void k_transpose_bf16(const float* __restrict__ W, __bf16* __restrict__ WT,
                                 int K, int N) {
    int id = blockIdx.x * blockDim.x + threadIdx.x;
    if (id >= K * N) return;
    int n = id / K;
    int k = id - n * K;
    WT[id] = (__bf16)W[(size_t)k * N + n];
}

// ------------------------------- WMMA GEMM ---------------------------------
// C[M,N] = A[M,K](bf16, row-major) * WT[N,K](bf16, N-major) + bias
// Block tile 128x128, BK=32, 4 waves (128 thr), 64x64 wave tiles.
// Global->LDS via async b128 copies into ping-pong buffers; wmma from LDS
// overlaps the next tile's async copies.
// EPI 0: split epilogue -> out0 = bf16 x_proj (cols<D), out1 = bf16 sigmoid gate
// EPI 1: out0 = fp32 (row stride N)

union FragU { v16bf v; struct { uint4 lo, hi; } q; };

template <int EPI>
__global__ __launch_bounds__(128)
void k_gemm_bf16(const __bf16* __restrict__ A, const __bf16* __restrict__ WT,
                 const float* __restrict__ bias, int M, int N, int K,
                 void* __restrict__ out0, void* __restrict__ out1) {
    constexpr int BM = 128, BN = 128, BK = 32;
    constexpr int LDA = BK + 8;   // 40 halves = 80B rows: 16B-aligned, bank-spread
    constexpr int ABUF = BM * LDA;
    __shared__ __bf16 As[2 * ABUF];
    __shared__ __bf16 Bs[2 * ABUF];

    const int tid  = threadIdx.x;
    const int lane = tid & 31;
    const int wave = tid >> 5;       // 0..3
    const int wm   = wave >> 1;      // 0..1 -> 64 rows
    const int wn   = wave & 1;       // 0..1 -> 64 cols
    const int bm   = blockIdx.y * BM;
    const int bn   = blockIdx.x * BN;
    const int hl   = lane >> 4;      // half of wave
    const int l16  = lane & 15;

    // Per-thread staging geometry: 512 x 16B chunks per operand, 4 per thread.
    const int r0  = tid >> 2;              // row for chunk i=0 (others +32)
    const int cc0 = (tid & 3) * 8;         // half-offset within the 32-wide row

    v8f acc[4][4];
#pragma unroll
    for (int mi = 0; mi < 4; ++mi)
#pragma unroll
        for (int ni = 0; ni < 4; ++ni) acc[mi][ni] = (v8f)(0.f);

    // Prologue: async-stage tile 0 into buffer 0 (8 async b128 per thread).
#pragma unroll
    for (int i = 0; i < 4; ++i) {
        int r = r0 + i * 32;
        ASYNC_CP16(As + r * LDA + cc0, A  + (size_t)(bm + r) * K + cc0);
        ASYNC_CP16(Bs + r * LDA + cc0, WT + (size_t)(bn + r) * K + cc0);
    }

    const int nk = K / BK;
    for (int it = 0; it < nk; ++it) {
        const __bf16* Ac = As + (it & 1) * ABUF;
        const __bf16* Bc = Bs + (it & 1) * ABUF;

        if (it + 1 < nk) {
            // Stage the next tile into the other buffer (safe: that buffer was
            // last read in iteration it-1, sealed by its trailing barrier).
            __bf16* An = As + ((it + 1) & 1) * ABUF;
            __bf16* Bn = Bs + ((it + 1) & 1) * ABUF;
            int kk = (it + 1) * BK;
#pragma unroll
            for (int i = 0; i < 4; ++i) {
                int r = r0 + i * 32;
                ASYNC_CP16(An + r * LDA + cc0, A  + (size_t)(bm + r) * K + kk + cc0);
                ASYNC_CP16(Bn + r * LDA + cc0, WT + (size_t)(bn + r) * K + kk + cc0);
            }
            // In-order completion: <=8 outstanding means current tile landed.
            WAIT_ASYNC_8();
        } else {
            WAIT_ASYNC_0();
        }
        // Warm L2 two tiles ahead.
        if (it + 2 < nk) {
            __builtin_prefetch(A  + (size_t)(bm + tid) * K + (it + 2) * BK, 0, 1);
            __builtin_prefetch(WT + (size_t)(bn + tid) * K + (it + 2) * BK, 0, 1);
        }
        __syncthreads();

        // B fragments: 32x16 tile; lanes<16 hold K=0..15 of col, lanes>=16
        // hold K=16..31; Bs is [N][K] so the 16 K-values are contiguous.
        v16bf bfr[4];
#pragma unroll
        for (int ni = 0; ni < 4; ++ni) {
            int col = wn * 64 + ni * 16 + l16;
            FragU u;
            u.q.lo = *(const uint4*)(Bc + col * LDA + hl * 16);
            u.q.hi = *(const uint4*)(Bc + col * LDA + hl * 16 + 8);
            bfr[ni] = u.v;
        }
        // A fragments: 16x32; lanes<16 hold K=0..15, lanes>=16 hold
        // K=16..31, row = lane&15.
#pragma unroll
        for (int mi = 0; mi < 4; ++mi) {
            int row = wm * 64 + mi * 16 + l16;
            FragU u;
            u.q.lo = *(const uint4*)(Ac + row * LDA + hl * 16);
            u.q.hi = *(const uint4*)(Ac + row * LDA + hl * 16 + 8);
            v16bf af = u.v;
#pragma unroll
            for (int ni = 0; ni < 4; ++ni)
                acc[mi][ni] = __builtin_amdgcn_wmma_f32_16x16x32_bf16(
                    false, af, false, bfr[ni], (short)0, acc[mi][ni], false, false);
        }
        __syncthreads();   // all waves done reading this buffer
    }

    // Epilogue. C layout: VGPR i -> M = i + 8*half, N = lane&15 (per tile).
#pragma unroll
    for (int mi = 0; mi < 4; ++mi) {
#pragma unroll
        for (int ni = 0; ni < 4; ++ni) {
            int colg = bn + wn * 64 + ni * 16 + l16;
            float bv = bias[colg];
#pragma unroll
            for (int i = 0; i < 8; ++i) {
                int rowg = bm + wm * 64 + mi * 16 + i + hl * 8;
                float v = acc[mi][ni][i] + bv;
                if (EPI == 0) {
                    if (colg < DIMD) {
                        ((__bf16*)out0)[(size_t)rowg * DIMD + colg] = (__bf16)v;
                    } else {
                        ((__bf16*)out1)[(size_t)rowg * DIMD + (colg - DIMD)] =
                            (__bf16)fast_sigmoid(v);
                    }
                } else {
                    ((float*)out0)[(size_t)rowg * N + colg] = v;
                }
            }
        }
    }
}

// --------------------------- dt low-rank stage ------------------------------
// dtl[m][r] = softplus( sum_k xp[m][k] * Wdt[k][r] + bdt[r] ), r<16.
__global__ __launch_bounds__(256)
void k_dtl(const __bf16* __restrict__ xp, const float* __restrict__ Wdt,
           const float* __restrict__ bdt, float* __restrict__ dtl) {
    int r   = threadIdx.x & 15;
    int lr  = threadIdx.x >> 4;
    int row = blockIdx.x * 16 + lr;
    const __bf16* xr = xp + (size_t)row * DIMD;
    float acc = 0.f;
    for (int k = 0; k < DIMD; k += 8) {
        v8bf xv = *(const v8bf*)(xr + k);
#pragma unroll
        for (int j = 0; j < 8; ++j)
            acc = fmaf((float)xv[j], Wdt[(k + j) * RNK + r], acc);
    }
    dtl[(size_t)row * RNK + r] = softplusf(acc + bdt[r]);
}

// ------------------------------ chunked scan --------------------------------
// Recurrence h = a*h + b with a = exp(dt*A_d), b = dt*B_t.

__device__ __forceinline__ float dt_dot(const float* __restrict__ dl,
                                        const float* __restrict__ w, float bd) {
    const float4* q = (const float4*)dl;
    float4 q0 = q[0], q1 = q[1], q2 = q[2], q3 = q[3];
    float dt = bd;
    dt = fmaf(q0.x, w[0], dt);  dt = fmaf(q0.y, w[1], dt);
    dt = fmaf(q0.z, w[2], dt);  dt = fmaf(q0.w, w[3], dt);
    dt = fmaf(q1.x, w[4], dt);  dt = fmaf(q1.y, w[5], dt);
    dt = fmaf(q1.z, w[6], dt);  dt = fmaf(q1.w, w[7], dt);
    dt = fmaf(q2.x, w[8], dt);  dt = fmaf(q2.y, w[9], dt);
    dt = fmaf(q2.z, w[10], dt); dt = fmaf(q2.w, w[11], dt);
    dt = fmaf(q3.x, w[12], dt); dt = fmaf(q3.y, w[13], dt);
    dt = fmaf(q3.z, w[14], dt); dt = fmaf(q3.w, w[15], dt);
    return dt;
}

// Pass 1: per (b,d,chunk) compute X = prod(a), Y = chunk-local solution.
__global__ __launch_bounds__(256)
void k_scan1(const float* __restrict__ dtl, const float* __restrict__ Bt,
             const float* __restrict__ Wdtp, const float* __restrict__ bdtp,
             const float* __restrict__ A_p,
             float* __restrict__ Xc, float* __restrict__ Yc) {
    int gid  = blockIdx.x * blockDim.x + threadIdx.x;  // ((b*NC+c)*D + d)
    int d    = gid & (DIMD - 1);
    int rest = gid >> 10;
    int c    = rest & (NCHUNK - 1);
    int b    = rest >> 4;

    float A = -softplusf(A_p[d]);
    float w[RNK];
#pragma unroll
    for (int r = 0; r < RNK; ++r) w[r] = Wdtp[r * DIMD + d];
    float bd = bdtp[d];

    float X = 1.f, Y = 0.f;
    size_t row0 = (size_t)b * SEQ + (size_t)c * CHLEN;
    for (int t = 0; t < CHLEN; ++t) {
        size_t row = row0 + t;
        float dt = dt_dot(dtl + row * RNK, w, bd);
        float a  = __expf(dt * A);
        float bb = dt * Bt[row * DIMD + d];
        X *= a;
        Y = fmaf(a, Y, bb);
    }
    Xc[gid] = X;
    Yc[gid] = Y;
}

// Pass 2: tiny serial scan over the 16 chunks -> chunk-entry states h0.
__global__ void k_scan2(const float* __restrict__ Xc, const float* __restrict__ Yc,
                        float* __restrict__ h0) {
    int gid = blockIdx.x * blockDim.x + threadIdx.x;  // b*D + d
    int d = gid & (DIMD - 1);
    int b = gid >> 10;
    float h = 0.f;
    for (int c = 0; c < NCHUNK; ++c) {
        size_t idx = ((size_t)(b * NCHUNK + c)) * DIMD + d;
        h0[idx] = h;
        h = fmaf(Xc[idx], h, Yc[idx]);
    }
}

// Pass 3: replay each chunk with known entry state; fuse gate; emit bf16 y.
__global__ __launch_bounds__(256)
void k_scan3(const float* __restrict__ dtl, const float* __restrict__ Bt,
             const float* __restrict__ Ct, const __bf16* __restrict__ gate,
             const float* __restrict__ Wdtp, const float* __restrict__ bdtp,
             const float* __restrict__ A_p, const float* __restrict__ h0,
             __bf16* __restrict__ y) {
    int gid  = blockIdx.x * blockDim.x + threadIdx.x;
    int d    = gid & (DIMD - 1);
    int rest = gid >> 10;
    int c    = rest & (NCHUNK - 1);
    int b    = rest >> 4;

    float A = -softplusf(A_p[d]);
    float w[RNK];
#pragma unroll
    for (int r = 0; r < RNK; ++r) w[r] = Wdtp[r * DIMD + d];
    float bd = bdtp[d];

    float h = h0[gid];
    size_t row0 = (size_t)b * SEQ + (size_t)c * CHLEN;
    for (int t = 0; t < CHLEN; ++t) {
        size_t row = row0 + t;
        float dt = dt_dot(dtl + row * RNK, w, bd);
        float a  = __expf(dt * A);
        float bb = dt * Bt[row * DIMD + d];
        h = fmaf(a, h, bb);
        float yv = Ct[row * DIMD + d] * h * (float)gate[row * DIMD + d];
        y[row * DIMD + d] = (__bf16)yv;
    }
}

// ------------------------------- launcher -----------------------------------

extern "C" void kernel_launch(void* const* d_in, const int* in_sizes, int n_in,
                              void* d_out, int out_size, void* d_ws, size_t ws_size,
                              hipStream_t stream) {
    const float* x    = (const float*)d_in[0];
    const float* Win  = (const float*)d_in[1];
    const float* bin_ = (const float*)d_in[2];
    const float* A_p  = (const float*)d_in[3];
    const float* WB   = (const float*)d_in[4];
    const float* bB   = (const float*)d_in[5];
    const float* WC   = (const float*)d_in[6];
    const float* bC   = (const float*)d_in[7];
    const float* Wdt  = (const float*)d_in[8];
    const float* bdt  = (const float*)d_in[9];
    const float* Wdtp = (const float*)d_in[10];
    const float* bdtp = (const float*)d_in[11];
    const float* Wout = (const float*)d_in[12];
    const float* bout = (const float*)d_in[13];

    const int M = MROWS;
    char* p = (char*)d_ws;
    auto take = [&](size_t bytes) {
        char* r = p;
        p += (bytes + 255) & ~(size_t)255;
        return r;
    };
    __bf16* xb   = (__bf16*)take((size_t)M * DIMD * 2);
    __bf16* WinT = (__bf16*)take((size_t)2 * DIMD * DIMD * 2);
    __bf16* WBT  = (__bf16*)take((size_t)DIMD * DIMD * 2);
    __bf16* WCT  = (__bf16*)take((size_t)DIMD * DIMD * 2);
    __bf16* WoT  = (__bf16*)take((size_t)DIMD * DIMD * 2);
    __bf16* xp   = (__bf16*)take((size_t)M * DIMD * 2);
    __bf16* gate = (__bf16*)take((size_t)M * DIMD * 2);
    float*  Bt   = (float*)take((size_t)M * DIMD * 4);
    float*  Ct   = (float*)take((size_t)M * DIMD * 4);
    float*  dtl  = (float*)take((size_t)M * RNK * 4);
    float*  Xc   = (float*)take((size_t)BATCH * NCHUNK * DIMD * 4);
    float*  Yc   = (float*)take((size_t)BATCH * NCHUNK * DIMD * 4);
    float*  h0   = (float*)take((size_t)BATCH * NCHUNK * DIMD * 4);
    __bf16* yb   = (__bf16*)take((size_t)M * DIMD * 2);

    // 1) precision staging
    k_f32_to_bf16<<<dim3((M * DIMD / 4 + 255) / 256), 256, 0, stream>>>(x, xb, M * DIMD);
    k_transpose_bf16<<<dim3((DIMD * 2 * DIMD + 255) / 256), 256, 0, stream>>>(Win, WinT, DIMD, 2 * DIMD);
    k_transpose_bf16<<<dim3((DIMD * DIMD + 255) / 256), 256, 0, stream>>>(WB, WBT, DIMD, DIMD);
    k_transpose_bf16<<<dim3((DIMD * DIMD + 255) / 256), 256, 0, stream>>>(WC, WCT, DIMD, DIMD);
    k_transpose_bf16<<<dim3((DIMD * DIMD + 255) / 256), 256, 0, stream>>>(Wout, WoT, DIMD, DIMD);

    // 2) input projection + sigmoid gate (split epilogue)
    k_gemm_bf16<0><<<dim3(2 * DIMD / 128, M / 128), 128, 0, stream>>>(
        xb, WinT, bin_, M, 2 * DIMD, DIMD, xp, gate);

    // 3) B_t, C_t projections (fp32 outputs feed the recurrence)
    k_gemm_bf16<1><<<dim3(DIMD / 128, M / 128), 128, 0, stream>>>(
        xp, WBT, bB, M, DIMD, DIMD, Bt, nullptr);
    k_gemm_bf16<1><<<dim3(DIMD / 128, M / 128), 128, 0, stream>>>(
        xp, WCT, bC, M, DIMD, DIMD, Ct, nullptr);

    // 4) low-rank dt stage (softplus fused)
    k_dtl<<<dim3(M / 16), 256, 0, stream>>>(xp, Wdt, bdt, dtl);

    // 5) chunked parallel scan (dt projection + exp + gate fused)
    k_scan1<<<dim3(BATCH * NCHUNK * DIMD / 256), 256, 0, stream>>>(dtl, Bt, Wdtp, bdtp, A_p, Xc, Yc);
    k_scan2<<<dim3(BATCH * DIMD / 256), 256, 0, stream>>>(Xc, Yc, h0);
    k_scan3<<<dim3(BATCH * NCHUNK * DIMD / 256), 256, 0, stream>>>(dtl, Bt, Ct, gate, Wdtp, bdtp, A_p, h0, yb);

    // 6) output projection -> fp32 d_out
    k_gemm_bf16<1><<<dim3(DIMD / 128, M / 128), 128, 0, stream>>>(
        yb, WoT, bout, M, DIMD, DIMD, (float*)d_out, nullptr);
}
